// Sequence_28973849379580
// MI455X (gfx1250) — compile-verified, matrix operand order checked by reading, not run on hardware
//
#include <hip/hip_runtime.h>

// ---------------------------------------------------------------------------
// 2-layer LSTM (B=8192, T=2048, NS=51) as a persistent WMMA kernel.
// Each wave32 owns 16 batch rows for all T steps. Weights (f16, padded/
// transposed) + h/c state live in LDS. Per step:
//   gates1 = h1 @ Whh1^T                  (26x v_wmma_f32_16x16x32_f16)
//   elementwise: + x*Wih1col + bias, sigmoid/tanh, update c1/h1 (fp32 state)
//   gates2 = h1 @ Wih2^T + h2 @ Whh2^T    (52x v_wmma_f32_16x16x32_f16)
//   elementwise: update c2/h2, y = h2 . Wlin + blin -> out[b][t]
// B-fragment loads are software-pipelined one N-tile ahead; the x stream is
// double-buffered one timestep ahead so no global-load wait sits on the
// recurrence critical path.
// ---------------------------------------------------------------------------

#define NS      51
#define NG      204          // 4*NS
#define NGP     208          // padded gate width (13 tiles of 16)
#define KP      72           // padded K row stride in halves (144B, conflict-free)
#define NT      13           // N tiles of 16
#define WAVES   4
#define RPW     16           // batch rows per wave
#define RPB     (WAVES * RPW)

// LDS budget (must match carving in kernel)
#define COMMON_F (3 * NGP + 64)                    // bias1, bias2, wih1col, wlin
#define WAVE_F   (RPW * NGP + 3 * RPW * 52 + 16)   // gates + c1 + c2 + h2f32 + xbuf
#define TOT_F    (COMMON_F + WAVES * WAVE_F)
#define WT_H     (3 * NGP * KP)                    // 3 weight mats, [n][k] halves
#define WAVE_H   (2 * RPW * KP)                    // h1 + h2
#define TOT_H    (WT_H + WAVES * WAVE_H)
#define SMEM_BYTES ((size_t)TOT_F * 4 + (size_t)TOT_H * 2)   // = 204480

typedef __attribute__((ext_vector_type(16))) _Float16     v16h;
typedef __attribute__((ext_vector_type(8)))  float        v8f;
typedef __attribute__((ext_vector_type(4)))  unsigned int v4u;

union FragU { v16h v; v4u q[2]; };

// A/B fragment for v_wmma_f32_16x16x32_f16 (ISA 7.12.2 16-bit layout):
// lane L: row = L&15, half-group = (L>>4)*8; VGPR0-3 hold K = base..base+7,
// VGPR4-7 hold K = base+16..base+23  -> two 16B LDS loads per fragment.
__device__ __forceinline__ v16h load_frag16(const _Float16* row, int khalf) {
  FragU f;
  f.q[0] = *(const v4u*)(row + khalf);
  f.q[1] = *(const v4u*)(row + khalf + 16);
  return f.v;
}

__device__ __forceinline__ float sigf(float x) {
  return 1.0f / (1.0f + __expf(-x));
}
__device__ __forceinline__ float tanh_fast(float x) {
  float e = __expf(2.0f * x);
  return 1.0f - 2.0f / (e + 1.0f);
}

__global__ void lstm2_wmma_kernel(
    const float* __restrict__ x,
    const float* __restrict__ Wih1, const float* __restrict__ Whh1,
    const float* __restrict__ bih1, const float* __restrict__ bhh1,
    const float* __restrict__ Wih2, const float* __restrict__ Whh2,
    const float* __restrict__ bih2, const float* __restrict__ bhh2,
    const float* __restrict__ Wlin, const float* __restrict__ blin,
    float* __restrict__ out, int T)
{
  extern __shared__ unsigned char smem_raw[];

  // ---- LDS carving (floats first, then halves; all 16B aligned) ----
  float* fbase = (float*)smem_raw;
  float* bias1  = fbase;                 // [NGP]
  float* bias2  = bias1 + NGP;           // [NGP]
  float* wih1c  = bias2 + NGP;           // [NGP]
  float* wlin   = wih1c + NGP;           // [64]
  float* wavef0 = wlin + 64;             // per-wave float regions

  _Float16* hbase = (_Float16*)(fbase + TOT_F);
  _Float16* Wt    = hbase;               // [3][NGP][KP]  B-matrix: Wt[n][k]=W[n][k]
  _Float16* waveh0 = Wt + WT_H;

  const int tid  = threadIdx.x;
  const int wave = tid >> 5;
  const int lane = tid & 31;

  float* gates = wavef0 + wave * WAVE_F;     // [RPW][NGP]
  float* c1    = gates + RPW * NGP;          // [RPW][52]
  float* c2    = c1 + RPW * 52;
  float* h2f   = c2 + RPW * 52;              // fp32 copy of h2 for the output dot
  float* xbuf  = h2f + RPW * 52;             // [RPW]
  _Float16* h1 = waveh0 + wave * WAVE_H;     // [RPW][KP]
  _Float16* h2 = h1 + RPW * KP;

  // ---- one-time init: weights -> f16 LDS (padded), biases, state = 0 ----
  for (int i = tid; i < WT_H; i += blockDim.x) {
    int mat = i / (NGP * KP);
    int rem = i - mat * (NGP * KP);
    int n = rem / KP;
    int k = rem - n * KP;
    const float* src = (mat == 0) ? Whh1 : ((mat == 1) ? Wih2 : Whh2);
    float v = (n < NG && k < NS) ? src[n * NS + k] : 0.0f;
    Wt[i] = (_Float16)v;
  }
  for (int n = tid; n < NGP; n += blockDim.x) {
    bias1[n] = (n < NG) ? (bih1[n] + bhh1[n]) : 0.0f;
    bias2[n] = (n < NG) ? (bih2[n] + bhh2[n]) : 0.0f;
    wih1c[n] = (n < NG) ? Wih1[n] : 0.0f;
  }
  for (int j = tid; j < 64; j += blockDim.x) wlin[j] = (j < NS) ? Wlin[j] : 0.0f;
  for (int i = lane; i < RPW * KP; i += 32) { h1[i] = (_Float16)0.0f; h2[i] = (_Float16)0.0f; }
  for (int i = lane; i < RPW * 52; i += 32) { c1[i] = 0.0f; c2[i] = 0.0f; }
  __syncthreads();

  const int   rowbase = blockIdx.x * RPB + wave * RPW;
  const float blin_v  = blin[0];
  const int   mn      = lane & 15;          // M (for A) or N (for B/D) index
  const int   khalf   = (lane >> 4) << 3;   // K half-group base: 0 or 8
  const int   mrow    = (lane >> 4) << 3;   // D-layout M base: 0 or 8

  const _Float16* wt1 = Wt;                         // Whh1
  const _Float16* wti = Wt + (size_t)NGP * KP;      // Wih2
  const _Float16* wth = Wt + (size_t)2 * NGP * KP;  // Whh2

  // x stream: double-buffered one step ahead (load for t+1 issues at start of
  // step t; its wait lands a full step later, hidden behind the WMMA work).
  const float* xrow = x + (size_t)(rowbase + mn) * T;
  float xreg = (lane < RPW) ? xrow[0] : 0.0f;

  for (int t = 0; t < T; ++t) {
    if (lane < RPW) {
      xbuf[mn] = xreg;                      // stage x_t for the elementwise phase
      if (t + 1 < T) {
        xreg = xrow[t + 1];                 // issue next step's load now
        __builtin_prefetch(xrow + t + 17, 0, 0);  // global_prefetch_b8
      }
    }
    __builtin_amdgcn_wave_barrier();

    // ---- layer 1: gates = h1 @ Whh1^T  (26 WMMA, B frags pipelined) ----
    {
      const _Float16* ar = h1 + mn * KP;
      v16h a0 = load_frag16(ar, khalf);
      v16h a1 = load_frag16(ar, 32 + khalf);
      v16h b0 = load_frag16(wt1 + mn * KP, khalf);
      v16h b1 = load_frag16(wt1 + mn * KP, 32 + khalf);
#pragma unroll
      for (int nt = 0; nt < NT; ++nt) {
        const int nn = (nt + 1 < NT) ? (nt + 1) : 0;
        const _Float16* brn = wt1 + (nn * 16 + mn) * KP;
        v16h nb0 = load_frag16(brn, khalf);           // prefetch next tile
        v16h nb1 = load_frag16(brn, 32 + khalf);
        v8f acc = {};
        acc = __builtin_amdgcn_wmma_f32_16x16x32_f16(false, a0, false, b0, (short)0, acc, false, false);
        acc = __builtin_amdgcn_wmma_f32_16x16x32_f16(false, a1, false, b1, (short)0, acc, false, false);
        const int n = nt * 16 + mn;
#pragma unroll
        for (int r = 0; r < 8; ++r) gates[(mrow + r) * NGP + n] = acc[r];
        b0 = nb0; b1 = nb1;
      }
    }
    __builtin_amdgcn_wave_barrier();

    // ---- layer-1 elementwise (fp32): gate order [i,f,g,o] ----
    for (int idx = lane; idx < RPW * NS; idx += 32) {
      int m = idx / NS;
      int j = idx - m * NS;
      float xv = xbuf[m];
      float gi = gates[m * NGP + j]          + xv * wih1c[j]          + bias1[j];
      float gf = gates[m * NGP + NS + j]     + xv * wih1c[NS + j]     + bias1[NS + j];
      float gg = gates[m * NGP + 2 * NS + j] + xv * wih1c[2 * NS + j] + bias1[2 * NS + j];
      float go = gates[m * NGP + 3 * NS + j] + xv * wih1c[3 * NS + j] + bias1[3 * NS + j];
      float cn = sigf(gf) * c1[m * 52 + j] + sigf(gi) * tanh_fast(gg);
      float hn = sigf(go) * tanh_fast(cn);
      c1[m * 52 + j] = cn;
      h1[m * KP + j] = (_Float16)hn;        // pad cols [NS,KP) stay zero
    }
    __builtin_amdgcn_wave_barrier();

    // ---- layer 2: gates = h1 @ Wih2^T + h2 @ Whh2^T  (52 WMMA, pipelined) ----
    {
      const _Float16* a1r = h1 + mn * KP;
      const _Float16* a2r = h2 + mn * KP;
      v16h a1k0 = load_frag16(a1r, khalf);
      v16h a1k1 = load_frag16(a1r, 32 + khalf);
      v16h a2k0 = load_frag16(a2r, khalf);
      v16h a2k1 = load_frag16(a2r, 32 + khalf);
      v16h bi0 = load_frag16(wti + mn * KP, khalf);
      v16h bi1 = load_frag16(wti + mn * KP, 32 + khalf);
      v16h bh0 = load_frag16(wth + mn * KP, khalf);
      v16h bh1 = load_frag16(wth + mn * KP, 32 + khalf);
#pragma unroll
      for (int nt = 0; nt < NT; ++nt) {
        const int nn = (nt + 1 < NT) ? (nt + 1) : 0;
        const _Float16* bin = wti + (nn * 16 + mn) * KP;
        const _Float16* bhn = wth + (nn * 16 + mn) * KP;
        v16h nbi0 = load_frag16(bin, khalf);          // prefetch next tile
        v16h nbi1 = load_frag16(bin, 32 + khalf);
        v16h nbh0 = load_frag16(bhn, khalf);
        v16h nbh1 = load_frag16(bhn, 32 + khalf);
        v8f acc = {};
        acc = __builtin_amdgcn_wmma_f32_16x16x32_f16(false, a1k0, false, bi0, (short)0, acc, false, false);
        acc = __builtin_amdgcn_wmma_f32_16x16x32_f16(false, a1k1, false, bi1, (short)0, acc, false, false);
        acc = __builtin_amdgcn_wmma_f32_16x16x32_f16(false, a2k0, false, bh0, (short)0, acc, false, false);
        acc = __builtin_amdgcn_wmma_f32_16x16x32_f16(false, a2k1, false, bh1, (short)0, acc, false, false);
        const int n = nt * 16 + mn;
#pragma unroll
        for (int r = 0; r < 8; ++r) gates[(mrow + r) * NGP + n] = acc[r];
        bi0 = nbi0; bi1 = nbi1; bh0 = nbh0; bh1 = nbh1;
      }
    }
    __builtin_amdgcn_wave_barrier();

    // ---- layer-2 elementwise ----
    for (int idx = lane; idx < RPW * NS; idx += 32) {
      int m = idx / NS;
      int j = idx - m * NS;
      float gi = gates[m * NGP + j]          + bias2[j];
      float gf = gates[m * NGP + NS + j]     + bias2[NS + j];
      float gg = gates[m * NGP + 2 * NS + j] + bias2[2 * NS + j];
      float go = gates[m * NGP + 3 * NS + j] + bias2[3 * NS + j];
      float cn = sigf(gf) * c2[m * 52 + j] + sigf(gi) * tanh_fast(gg);
      float hn = sigf(go) * tanh_fast(cn);
      c2[m * 52 + j] = cn;
      h2[m * KP + j] = (_Float16)hn;
      h2f[m * 52 + j] = hn;
    }
    __builtin_amdgcn_wave_barrier();

    // ---- output: y[b][t] = h2 . Wlin + blin ----
    if (lane < RPW) {
      float acc = blin_v;
#pragma unroll
      for (int j = 0; j < NS; ++j) acc += h2f[mn * 52 + j] * wlin[j];
      out[(size_t)(rowbase + mn) * T + t] = acc;
    }
    __builtin_amdgcn_wave_barrier();
  }
}

extern "C" void kernel_launch(void* const* d_in, const int* in_sizes, int n_in,
                              void* d_out, int out_size, void* d_ws, size_t ws_size,
                              hipStream_t stream) {
  (void)n_in; (void)out_size; (void)d_ws; (void)ws_size;
  const float* x    = (const float*)d_in[0];
  const float* Wih1 = (const float*)d_in[1];
  const float* Whh1 = (const float*)d_in[2];
  const float* bih1 = (const float*)d_in[3];
  const float* bhh1 = (const float*)d_in[4];
  const float* Wih2 = (const float*)d_in[5];
  const float* Whh2 = (const float*)d_in[6];
  const float* bih2 = (const float*)d_in[7];
  const float* bhh2 = (const float*)d_in[8];
  const float* Wlin = (const float*)d_in[9];
  const float* blin = (const float*)d_in[10];
  float* out = (float*)d_out;

  const int T = 2048;                 // reference: x is [B, T, 1]
  const int B = in_sizes[0] / T;      // 8192

  // >64KB dynamic LDS (gfx1250 WGP supports 320KB/workgroup)
  hipFuncSetAttribute((const void*)lstm2_wmma_kernel,
                      hipFuncAttributeMaxDynamicSharedMemorySize,
                      (int)SMEM_BYTES);

  dim3 grid(B / RPB);   // 128 blocks of 4 waves (16 rows each)
  dim3 block(WAVES * 32);
  lstm2_wmma_kernel<<<grid, block, SMEM_BYTES, stream>>>(
      x, Wih1, Whh1, bih1, bhh1, Wih2, Whh2, bih2, bhh2, Wlin, blin, out, T);
}
